// ExpertsFeedForward_45028437131470
// MI455X (gfx1250) — matrix operand order repaired and checked by reading, not compiled, not run on hardware
//
#include <hip/hip_runtime.h>
#include <hip/hip_bf16.h>
#include <stdint.h>

// ---------------- problem constants ----------------
#define NTOK    16384   // B*S
#define DDIM    1024
#define HDIM    4096
#define NEXP    8
#define NFF     7
#define CAP     1638
#define CAP_PAD 1664    // CAP rounded up to multiple of 128

// ---------------- CDNA5 WMMA types ----------------
typedef __bf16 v16bf __attribute__((ext_vector_type(16)));
typedef float  v8f   __attribute__((ext_vector_type(8)));

union BF16Frag { unsigned int u[8]; v16bf v; };

// Native f32 -> bf16 via __bf16 cast (hardware cvt, not VALU emulation)
__device__ __forceinline__ unsigned short bf1(float a) {
  __bf16 h = (__bf16)a;
  return __builtin_bit_cast(unsigned short, h);
}
__device__ __forceinline__ unsigned int pk_bf16(float a, float b) {
  return (unsigned int)bf1(a) | ((unsigned int)bf1(b) << 16);
}

__device__ __forceinline__ float gelu_tanh(float x) {
  const float k = 0.7978845608028654f;
  float x3 = x * x * x;
  return 0.5f * x * (1.0f + tanhf(k * (x + 0.044715f * x3)));
}

// =========================================================================
// 1) Gating: one wave per token. logits = x@gate_W + b, softmax(l/T).
//    probs written transposed: probs[e*NTOK + t]
// =========================================================================
__global__ void __launch_bounds__(256)
gate_kernel(const float* __restrict__ x, const float* __restrict__ gW,
            const float* __restrict__ gb, const float* __restrict__ temp,
            float* __restrict__ probs)
{
  const int lane = threadIdx.x & 31;
  const int wv   = threadIdx.x >> 5;
  const int tok  = blockIdx.x * 8 + wv;

  float acc[NEXP];
  #pragma unroll
  for (int e = 0; e < NEXP; ++e) acc[e] = 0.f;

  for (int d = lane; d < DDIM; d += 32) {
    float xv = x[(size_t)tok * DDIM + d];
    float4 w0 = *reinterpret_cast<const float4*>(gW + (size_t)d * NEXP);
    float4 w1 = *reinterpret_cast<const float4*>(gW + (size_t)d * NEXP + 4);
    acc[0] += xv * w0.x; acc[1] += xv * w0.y; acc[2] += xv * w0.z; acc[3] += xv * w0.w;
    acc[4] += xv * w1.x; acc[5] += xv * w1.y; acc[6] += xv * w1.z; acc[7] += xv * w1.w;
  }
  #pragma unroll
  for (int off = 16; off > 0; off >>= 1) {
    #pragma unroll
    for (int e = 0; e < NEXP; ++e) acc[e] += __shfl_xor(acc[e], off, 32);
  }
  if (lane == 0) {
    float t = fmaxf(temp[0], 0.1f);
    float l[NEXP], mx = -1e30f;
    #pragma unroll
    for (int e = 0; e < NEXP; ++e) { l[e] = (acc[e] + gb[e]) / t; mx = fmaxf(mx, l[e]); }
    float s = 0.f;
    #pragma unroll
    for (int e = 0; e < NEXP; ++e) { l[e] = expf(l[e] - mx); s += l[e]; }
    float inv = 1.f / s;
    #pragma unroll
    for (int e = 0; e < NEXP; ++e) probs[(size_t)e * NTOK + tok] = l[e] * inv;
  }
}

// =========================================================================
// 2) Per-expert top-CAP selection via bit-pattern histogram (probs >= 0 so
//    float bits are order-preserving). One block per expert.
// =========================================================================
__global__ void __launch_bounds__(1024)
topk_kernel(const float* __restrict__ probs, int* __restrict__ oidx,
            float* __restrict__ oscore)
{
  __shared__ unsigned int hist[8192];
  __shared__ int sT, sBase, sNeed;
  __shared__ unsigned int cHigh, cTie;

  const int tid = threadIdx.x;
  const float* p = probs + (size_t)blockIdx.x * NTOK;
  int*   ei = oidx   + (size_t)blockIdx.x * CAP_PAD;
  float* es = oscore + (size_t)blockIdx.x * CAP_PAD;

  for (int i = tid; i < 8192; i += 1024) hist[i] = 0;
  if (tid == 0) { cHigh = 0; cTie = 0; }
  __syncthreads();

  for (int t = tid; t < NTOK; t += 1024) {
    unsigned b = (__float_as_uint(p[t]) >> 17) & 8191u;
    atomicAdd(&hist[b], 1u);
  }
  __syncthreads();

  if (tid == 0) {
    unsigned cum = 0; int T = 0;
    for (int b = 8191; b >= 0; --b) {
      unsigned nc = cum + hist[b];
      if (nc >= (unsigned)CAP) { T = b; break; }
      cum = nc;
    }
    sT = T; sBase = (int)cum; sNeed = CAP - (int)cum;
  }
  __syncthreads();

  for (int i = CAP + tid; i < CAP_PAD; i += 1024) { ei[i] = -1; es[i] = 0.f; }

  const int T = sT, base = sBase, need = sNeed;
  for (int t = tid; t < NTOK; t += 1024) {
    float v = p[t];
    int b = (int)((__float_as_uint(v) >> 17) & 8191u);
    if (b > T) {
      unsigned pos = atomicAdd(&cHigh, 1u);
      ei[pos] = t; es[pos] = v;
    } else if (b == T) {
      unsigned q = atomicAdd(&cTie, 1u);
      if ((int)q < need) { ei[base + q] = t; es[base + q] = v; }
    }
  }
}

// =========================================================================
// 3) GEMM #1: H = gelu(A@W + bias), A f32 (optionally row-gathered),
//    W f32 (KxN), output bf16. 128x128x32 tiles, 8 waves, bf16 WMMA.
//    Ping-pong LDS double buffering: one barrier per K-step, global loads
//    of tile k+1 overlap WMMA on tile k; prefetch of tile k+2 into L2.
// =========================================================================
__global__ void __launch_bounds__(256)
gemm_bf16_gelu_k1(const float* __restrict__ A, int lda,
                  const float* __restrict__ W, int ldw,
                  const float* __restrict__ bias,
                  const int* __restrict__ gidx,
                  unsigned short* __restrict__ Hout, int ldh,
                  int K)
{
  __shared__ unsigned short lA[2][128][40];   // [buf][m][k] bf16
  __shared__ unsigned short lB[2][128][40];   // [buf][n][k] bf16 (k innermost)

  const int tid   = threadIdx.x;
  const int lane  = tid & 31, wv = tid >> 5;
  const int waveM = wv & 3,  waveN = wv >> 2;
  const int mBlk  = blockIdx.x * 128, nBlk = blockIdx.y * 128;

  // per-thread loader coordinates
  const int arow = tid >> 1,        ac4 = (tid & 1) * 4;   // A: 2 float4/row pair
  const int brow = tid >> 3,        bc4 = (tid & 7) * 4;   // (unused naming kept simple below)

  v8f zero = {0.f,0.f,0.f,0.f,0.f,0.f,0.f,0.f};
  v8f acc[2][4];
  #pragma unroll
  for (int i = 0; i < 2; ++i)
    #pragma unroll
    for (int j = 0; j < 4; ++j) acc[i][j] = zero;

  float4 af[4], bfv[4];

  // ---- register-tile loaders ----
  auto loadA = [&](int k0) {
    #pragma unroll
    for (int t = 0; t < 4; ++t) {
      int v = tid + t * 256;
      int row = v >> 3, c4 = v & 7;
      int gRow = mBlk + row;
      int sRow = gidx ? gidx[gRow] : gRow;
      float4 f = make_float4(0.f, 0.f, 0.f, 0.f);
      if (sRow >= 0)
        f = *reinterpret_cast<const float4*>(A + (size_t)sRow * lda + k0 + c4 * 4);
      af[t] = f;
    }
  };
  auto loadB = [&](int k0) {
    #pragma unroll
    for (int t = 0; t < 4; ++t) {
      int v = tid + t * 256;
      int k = v >> 5, c4 = v & 31;
      bfv[t] = *reinterpret_cast<const float4*>(W + (size_t)(k0 + k) * ldw + nBlk + c4 * 4);
    }
  };
  auto storeA = [&](int buf) {
    #pragma unroll
    for (int t = 0; t < 4; ++t) {
      int v = tid + t * 256;
      int row = v >> 3, c4 = v & 7;
      unsigned int* d = reinterpret_cast<unsigned int*>(&lA[buf][row][c4 * 4]);
      d[0] = pk_bf16(af[t].x, af[t].y);
      d[1] = pk_bf16(af[t].z, af[t].w);
    }
  };
  auto storeB = [&](int buf) {
    #pragma unroll
    for (int t = 0; t < 4; ++t) {
      int v = tid + t * 256;
      int k = v >> 5, c4 = v & 31;
      lB[buf][c4 * 4 + 0][k] = bf1(bfv[t].x);
      lB[buf][c4 * 4 + 1][k] = bf1(bfv[t].y);
      lB[buf][c4 * 4 + 2][k] = bf1(bfv[t].z);
      lB[buf][c4 * 4 + 3][k] = bf1(bfv[t].w);
    }
  };

  // ---- prologue: tile 0 into buffer 0 ----
  loadA(0); loadB(0);
  storeA(0); storeB(0);
  __syncthreads();

  const int ml  = lane & 15;
  const int kbA = (lane >> 4) * 16;      // A 16x32: contiguous k = 16*(lane/16) + s
  const int kbB = (lane >> 4) * 16;      // B 32x16 layout

  int buf = 0;
  for (int k0 = 0; k0 < K; k0 += 32, buf ^= 1) {
    const bool more = (k0 + 32) < K;
    if (more) {                           // overlap next tile loads with WMMA
      loadA(k0 + 32);
      loadB(k0 + 32);
    }
    if (k0 + 64 < K) {                    // L2 prefetch two tiles ahead
      __builtin_prefetch(W + (size_t)(k0 + 64 + (tid >> 3)) * ldw + nBlk + (tid & 7) * 16, 0, 3);
    }

    BF16Frag fa[2];
    #pragma unroll
    for (int mi = 0; mi < 2; ++mi) {
      int m16 = waveM * 32 + mi * 16 + ml;
      #pragma unroll
      for (int p = 0; p < 8; ++p) {
        int k = 2 * p + kbA;
        fa[mi].u[p] = *reinterpret_cast<const unsigned int*>(&lA[buf][m16][k]);
      }
    }
    #pragma unroll
    for (int ni = 0; ni < 4; ++ni) {
      BF16Frag fb;
      int n16 = waveN * 64 + ni * 16 + ml;
      #pragma unroll
      for (int p = 0; p < 8; ++p)
        fb.u[p] = *reinterpret_cast<const unsigned int*>(&lB[buf][n16][kbB + 2 * p]);
      #pragma unroll
      for (int mi = 0; mi < 2; ++mi)
        acc[mi][ni] = __builtin_amdgcn_wmma_f32_16x16x32_bf16(
            false, fa[mi].v, false, fb.v, (short)0, acc[mi][ni], false, false);
    }

    if (more) {                           // fill the other buffer
      storeA(buf ^ 1);
      storeB(buf ^ 1);
    }
    __syncthreads();
  }

  // epilogue: gelu(acc+bias) -> bf16 Hout
  const int mOff = (lane >> 4) * 8;       // C/D: VGPR r -> M = r (+8 hi lanes)
  #pragma unroll
  for (int ni = 0; ni < 4; ++ni) {
    int n = nBlk + waveN * 64 + ni * 16 + ml;
    float b = bias[n];
    #pragma unroll
    for (int mi = 0; mi < 2; ++mi) {
      int mBase = mBlk + waveM * 32 + mi * 16 + mOff;
      #pragma unroll
      for (int r = 0; r < 8; ++r) {
        float vv = acc[mi][ni][r] + b;
        Hout[(size_t)(mBase + r) * ldh + n] = bf1(gelu_tanh(vv));
      }
    }
  }
}

// =========================================================================
// 4) GEMM #2: Out = H_bf16 @ W + bias. scatter==0: plain f32 store.
//    scatter==1: Out[gidx[m]] += val * gscore[m] (atomic, skip idx<0).
//    Same double-buffered pipeline; A is already bf16 (uint4 loads).
// =========================================================================
__global__ void __launch_bounds__(256)
gemm_bf16_out_k2(const unsigned short* __restrict__ Hin, int ldhin,
                 const float* __restrict__ W, int ldw,
                 const float* __restrict__ bias,
                 const int* __restrict__ gidx,
                 const float* __restrict__ gscore,
                 float* __restrict__ Out,
                 int K, int scatter)
{
  __shared__ unsigned short lA[2][128][40];
  __shared__ unsigned short lB[2][128][40];

  const int tid   = threadIdx.x;
  const int lane  = tid & 31, wv = tid >> 5;
  const int waveM = wv & 3,  waveN = wv >> 2;
  const int mBlk  = blockIdx.x * 128, nBlk = blockIdx.y * 128;

  v8f zero = {0.f,0.f,0.f,0.f,0.f,0.f,0.f,0.f};
  v8f acc[2][4];
  #pragma unroll
  for (int i = 0; i < 2; ++i)
    #pragma unroll
    for (int j = 0; j < 4; ++j) acc[i][j] = zero;

  uint4  av[2];
  float4 bfv[4];

  auto loadA = [&](int k0) {
    #pragma unroll
    for (int t = 0; t < 2; ++t) {
      int v = tid + t * 256;
      int row = v >> 2, c = v & 3;
      const unsigned short* rp = Hin + (size_t)(mBlk + row) * ldhin + k0;
      av[t] = reinterpret_cast<const uint4*>(rp)[c];
    }
  };
  auto loadB = [&](int k0) {
    #pragma unroll
    for (int t = 0; t < 4; ++t) {
      int v = tid + t * 256;
      int k = v >> 5, c4 = v & 31;
      bfv[t] = *reinterpret_cast<const float4*>(W + (size_t)(k0 + k) * ldw + nBlk + c4 * 4);
    }
  };
  auto storeA = [&](int buf) {
    #pragma unroll
    for (int t = 0; t < 2; ++t) {
      int v = tid + t * 256;
      int row = v >> 2, c = v & 3;
      unsigned int* dst = reinterpret_cast<unsigned int*>(&lA[buf][row][c * 8]);
      dst[0] = av[t].x; dst[1] = av[t].y; dst[2] = av[t].z; dst[3] = av[t].w;
    }
  };
  auto storeB = [&](int buf) {
    #pragma unroll
    for (int t = 0; t < 4; ++t) {
      int v = tid + t * 256;
      int k = v >> 5, c4 = v & 31;
      lB[buf][c4 * 4 + 0][k] = bf1(bfv[t].x);
      lB[buf][c4 * 4 + 1][k] = bf1(bfv[t].y);
      lB[buf][c4 * 4 + 2][k] = bf1(bfv[t].z);
      lB[buf][c4 * 4 + 3][k] = bf1(bfv[t].w);
    }
  };

  loadA(0); loadB(0);
  storeA(0); storeB(0);
  __syncthreads();

  const int ml  = lane & 15;
  const int kbA = (lane >> 4) * 16;
  const int kbB = (lane >> 4) * 16;

  int buf = 0;
  for (int k0 = 0; k0 < K; k0 += 32, buf ^= 1) {
    const bool more = (k0 + 32) < K;
    if (more) {
      loadA(k0 + 32);
      loadB(k0 + 32);
    }
    if (k0 + 64 < K) {
      __builtin_prefetch(W + (size_t)(k0 + 64 + (tid >> 3)) * ldw + nBlk + (tid & 7) * 16, 0, 3);
    }

    BF16Frag fa[2];
    #pragma unroll
    for (int mi = 0; mi < 2; ++mi) {
      int m16 = waveM * 32 + mi * 16 + ml;
      #pragma unroll
      for (int p = 0; p < 8; ++p) {
        int k = 2 * p + kbA;
        fa[mi].u[p] = *reinterpret_cast<const unsigned int*>(&lA[buf][m16][k]);
      }
    }
    #pragma unroll
    for (int ni = 0; ni < 4; ++ni) {
      BF16Frag fb;
      int n16 = waveN * 64 + ni * 16 + ml;
      #pragma unroll
      for (int p = 0; p < 8; ++p)
        fb.u[p] = *reinterpret_cast<const unsigned int*>(&lB[buf][n16][kbB + 2 * p]);
      #pragma unroll
      for (int mi = 0; mi < 2; ++mi)
        acc[mi][ni] = __builtin_amdgcn_wmma_f32_16x16x32_bf16(
            false, fa[mi].v, false, fb.v, (short)0, acc[mi][ni], false, false);
    }

    if (more) {
      storeA(buf ^ 1);
      storeB(buf ^ 1);
    }
    __syncthreads();
  }

  const int mOff = (lane >> 4) * 8;
  #pragma unroll
  for (int mi = 0; mi < 2; ++mi) {
    #pragma unroll
    for (int r = 0; r < 8; ++r) {
      int m = mBlk + waveM * 32 + mi * 16 + mOff + r;
      if (scatter) {
        int srow = gidx[m];
        if (srow >= 0) {
          float s = gscore[m];
          #pragma unroll
          for (int ni = 0; ni < 4; ++ni) {
            int n = nBlk + waveN * 64 + ni * 16 + ml;
            float vv = (acc[mi][ni][r] + bias[n]) * s;
            atomicAdd(&Out[(size_t)srow * ldw + n], vv);
          }
        }
      } else {
        #pragma unroll
        for (int ni = 0; ni < 4; ++ni) {
          int n = nBlk + waveN * 64 + ni * 16 + ml;
          Out[(size_t)m * ldw + n] = acc[mi][ni][r] + bias[n];
        }
      }
    }
  }
}

// =========================================================================
// 5) Jump expert: Out[idx[c]] += jump * score[c]
// =========================================================================
__global__ void __launch_bounds__(256)
jump_kernel(const int* __restrict__ idx, const float* __restrict__ score,
            const float* __restrict__ jmp, float* __restrict__ Out)
{
  int c = blockIdx.x;
  int row = idx[c];
  if (row < 0) return;
  float s = score[c];
  for (int d = threadIdx.x; d < DDIM; d += 256)
    atomicAdd(&Out[(size_t)row * DDIM + d], jmp[d] * s);
}

// =========================================================================
extern "C" void kernel_launch(void* const* d_in, const int* in_sizes, int n_in,
                              void* d_out, int out_size, void* d_ws, size_t ws_size,
                              hipStream_t stream)
{
  const float* x    = (const float*)d_in[0];
  const float* gW   = (const float*)d_in[1];
  const float* gb   = (const float*)d_in[2];
  const float* temp = (const float*)d_in[3];
  const float* Wk   = (const float*)d_in[4];
  const float* bk   = (const float*)d_in[5];
  const float* Wv   = (const float*)d_in[6];
  const float* bv   = (const float*)d_in[7];
  const float* jmp  = (const float*)d_in[8];
  const float* sWk  = (const float*)d_in[9];
  const float* sbk  = (const float*)d_in[10];
  const float* sWv  = (const float*)d_in[11];
  const float* sbv  = (const float*)d_in[12];
  float* out = (float*)d_out;

  char* ws = (char*)d_ws;
  float* probs  = (float*)ws;  ws += (size_t)NEXP * NTOK * sizeof(float);
  int*   tidx   = (int*)ws;    ws += (size_t)NEXP * CAP_PAD * sizeof(int);
  float* tscore = (float*)ws;  ws += (size_t)NEXP * CAP_PAD * sizeof(float);
  ws = (char*)(((uintptr_t)ws + 255) & ~(uintptr_t)255);
  unsigned short* Hbuf = (unsigned short*)ws;   // NTOK x HDIM bf16 (128 MB, fits L2)

  // gating + selection
  gate_kernel<<<NTOK / 8, 256, 0, stream>>>(x, gW, gb, temp, probs);
  topk_kernel<<<NEXP, 1024, 0, stream>>>(probs, tidx, tscore);

  // shared FFN: out = gelu(x@sWk + sbk) @ sWv + sbv
  gemm_bf16_gelu_k1<<<dim3(NTOK / 128, HDIM / 128), 256, 0, stream>>>(
      x, DDIM, sWk, HDIM, sbk, nullptr, Hbuf, HDIM, DDIM);
  gemm_bf16_out_k2<<<dim3(NTOK / 128, DDIM / 128), 256, 0, stream>>>(
      Hbuf, HDIM, sWv, DDIM, sbv, nullptr, nullptr, out, HDIM, 0);

  // expert FFNs (gather -> GEMM -> gelu -> GEMM -> scaled scatter-add)
  for (int e = 0; e < NFF; ++e) {
    gemm_bf16_gelu_k1<<<dim3(CAP_PAD / 128, HDIM / 128), 256, 0, stream>>>(
        x, DDIM, Wk + (size_t)e * DDIM * HDIM, HDIM, bk + (size_t)e * HDIM,
        tidx + e * CAP_PAD, Hbuf, HDIM, DDIM);
    gemm_bf16_out_k2<<<dim3(CAP_PAD / 128, DDIM / 128), 256, 0, stream>>>(
        Hbuf, HDIM, Wv + (size_t)e * HDIM * DDIM, DDIM, bv + (size_t)e * DDIM,
        tidx + e * CAP_PAD, tscore + e * CAP_PAD, out, HDIM, 1);
  }

  // jump expert
  jump_kernel<<<CAP_PAD, 256, 0, stream>>>(tidx + NFF * CAP_PAD,
                                           tscore + NFF * CAP_PAD, jmp, out);
}
